// SimpleAttention_39032662786671
// MI455X (gfx1250) — matrix operand order repaired and checked by reading, not compiled, not run on hardware
//
#include <hip/hip_runtime.h>

#define BDIM 128
#define SDIM 512
#define EDIM 256
#define UDIM 10
#define CDIM 32
#define UP   48   // 4U=40 padded to 48 (3 WMMA row tiles)
#define KP2  32   // layer-2 K: 2U=20 padded to 32 (one WMMA K step)
#define NKT  (SDIM / 32)   // 16 K-tiles in attention GEMM
#define NET  (EDIM / 16)   // 16 e row-tiles
#define NTT  (SDIM / 16)   // 32 t col-tiles

typedef __attribute__((ext_vector_type(16))) _Float16 v16h;
typedef __attribute__((ext_vector_type(8)))  float    v8f;

union Frag16 { v16h v; _Float16 h[16]; };

// A fragment 16x32 (MxK), row-major source with stride lda (used by k_xw).
// ISA 7.12.2: lanes 0-15 hold M=lane, K = {0..7, 16..23}; lanes 16-31 K = {8..15, 24..31}.
__device__ inline v16h load_a_frag(const _Float16* __restrict__ A, int lda, int lane) {
  Frag16 u;
  const int m  = lane & 15;
  const int kb = (lane >> 4) * 8;
  const _Float16* r = A + (size_t)m * lda;
#pragma unroll
  for (int j = 0; j < 8; ++j) {
    u.h[j]     = r[kb + j];
    u.h[8 + j] = r[16 + kb + j];
  }
  return u.v;
}

// B fragment 32x16 (KxN) from an N-major source Bt[N][K] (used by k_xw).
__device__ inline v16h load_bT_frag(const _Float16* __restrict__ Bt, int ldk, int lane) {
  Frag16 u;
  const int n  = lane & 15;
  const int kb = (lane >> 4) * 8;
  const _Float16* p = Bt + (size_t)n * ldk + kb;
#pragma unroll
  for (int j = 0; j < 16; ++j) u.h[j] = p[(j < 8) ? j : (j + 8)];
  return u.v;
}

// ---------------- prep kernels ----------------

// Pack W1 into B-fragment-linear order: w1P[tt][kt][lane][j], one aligned v16h per lane.
// Element (k = kt*32 + (j<8 ? (lane>>4)*8 + j : 16 + (lane>>4)*8 + (j-8)), col t = tt*16 + (lane&15)) = W1[k][t].
__global__ void k_pack_w1(const float* __restrict__ W1, _Float16* __restrict__ w1P) {
  int i = blockIdx.x * blockDim.x + threadIdx.x;
  if (i >= NTT * NKT * 32 * 16) return;
  int j    = i & 15;
  int lane = (i >> 4) & 31;
  int kt   = (i >> 9) & (NKT - 1);
  int tt   = i >> 13;
  int k = kt * 32 + ((j < 8) ? ((lane >> 4) * 8 + j) : (16 + (lane >> 4) * 8 + (j - 8)));
  int t = tt * 16 + (lane & 15);
  w1P[i] = (_Float16)W1[(size_t)k * SDIM + t];
}

// Pack inputs into A-fragment-linear order: atP[b][et][kt][lane][j].
// Lane m = lane&15 -> row e = et*16+m; j<8 -> K = kt*32 + (lane>>4)*8 + j,
// j>=8 -> K = kt*32 + 16 + (lane>>4)*8 + (j-8); K indexes s.
__global__ void k_pack_a(const float* __restrict__ in, _Float16* __restrict__ atP) {
  size_t i = (size_t)blockIdx.x * blockDim.x + threadIdx.x;
  if (i >= (size_t)BDIM * NET * NKT * 32 * 16) return;
  int j    = (int)(i & 15);
  int lane = (int)((i >> 4) & 31);
  int kt   = (int)((i >> 9) & (NKT - 1));
  int et   = (int)((i >> 13) & (NET - 1));
  int b    = (int)(i >> 17);
  int kb   = (lane >> 4) * 8;
  int s    = kt * 32 + ((j < 8) ? (kb + j) : (16 + kb + (j - 8)));
  int e    = et * 16 + (lane & 15);
  atP[i] = (_Float16)in[((size_t)b * SDIM + s) * EDIM + e];
}

__global__ void k_pack_k(const float* __restrict__ k1f, const float* __restrict__ k1b,
                         const float* __restrict__ k2f, const float* __restrict__ k2b,
                         _Float16* __restrict__ k1T, _Float16* __restrict__ k2T) {
  int i = blockIdx.x * blockDim.x + threadIdx.x;
  const int n1 = 2 * UP * EDIM;
  const int n2 = 2 * UP * KP2;
  if (i < n1) {
    int dir = i / (UP * EDIM);
    int r   = i % (UP * EDIM);
    int j = r / EDIM, e = r % EDIM;
    const float* k = dir ? k1b : k1f;
    k1T[i] = (_Float16)((j < 4 * UDIM) ? k[e * 4 * UDIM + j] : 0.f);
  } else if (i < n1 + n2) {
    int r2  = i - n1;
    int dir = r2 / (UP * KP2);
    int r   = r2 % (UP * KP2);
    int j = r / KP2, u = r % KP2;
    const float* k = dir ? k2b : k2f;
    k2T[r2] = (_Float16)((j < 4 * UDIM && u < 2 * UDIM) ? k[u * 4 * UDIM + j] : 0.f);
  }
}

__global__ void k_zero_h1(_Float16* __restrict__ h1S) {
  size_t i = (size_t)blockIdx.x * blockDim.x + threadIdx.x;
  if (i < (size_t)BDIM * SDIM * KP2) h1S[i] = (_Float16)0.f;
}

// ---------------- fused attention: WMMA GEMM + bias + softmax + gate ----------------
// Block = (b, 16 e-rows); 8 waves x 4 col tiles = 512 columns.
// Packed operands -> every fragment is one aligned 32B load; K-loop unrolled x2
// with ping-pong fragment registers (no rotate moves).

__global__ __launch_bounds__(256) void k_attn(
    const float* __restrict__ inputs, const _Float16* __restrict__ atP,
    const _Float16* __restrict__ w1P, const float* __restrict__ b1,
    _Float16* __restrict__ gatedS) {
  __shared__ float red[8][16];
  const int b    = blockIdx.y;
  const int et   = blockIdx.x;
  const int e0   = et * 16;
  const int wave = threadIdx.x >> 5;
  const int lane = threadIdx.x & 31;
  const int n    = lane & 15;
  const int hh   = lane >> 4;

  const _Float16* Ap = atP + ((((size_t)b * NET + et) * NKT) * 32 + lane) * 16;
  const _Float16* Bp = w1P + (size_t)lane * 16;

  auto loadA = [&](int kt) -> v16h {
    return *(const v16h*)(Ap + (size_t)kt * (32 * 16));
  };
  auto loadB = [&](int ct, int kt) -> v16h {
    return *(const v16h*)(Bp + ((size_t)ct * NKT + kt) * (32 * 16));
  };

  v8f acc[4];
#pragma unroll
  for (int i = 0; i < 4; ++i)
#pragma unroll
    for (int v = 0; v < 8; ++v) acc[i][v] = 0.f;

  // prologue: kt = 0 fragments
  v16h a0 = loadA(0);
  v16h b0[4];
#pragma unroll
  for (int i = 0; i < 4; ++i) b0[i] = loadB(wave * 4 + i, 0);

  for (int kt = 0; kt < NKT; kt += 2) {
    // odd buffer (kt+1 is always valid: NKT even)
    v16h a1 = loadA(kt + 1);
    v16h b1f[4];
#pragma unroll
    for (int i = 0; i < 4; ++i) b1f[i] = loadB(wave * 4 + i, kt + 1);
#pragma unroll
    for (int i = 0; i < 4; ++i)
      acc[i] = __builtin_amdgcn_wmma_f32_16x16x32_f16(false, a0, false, b0[i],
                                                      (short)0, acc[i], false, false);
    // even buffer for next pair (clamped on last iteration -> cache-hot reload)
    const int ktn = (kt + 2 < NKT) ? (kt + 2) : kt;
    a0 = loadA(ktn);
#pragma unroll
    for (int i = 0; i < 4; ++i) b0[i] = loadB(wave * 4 + i, ktn);
#pragma unroll
    for (int i = 0; i < 4; ++i)
      acc[i] = __builtin_amdgcn_wmma_f32_16x16x32_f16(false, a1, false, b1f[i],
                                                      (short)0, acc[i], false, false);
  }

  // bias over column t
#pragma unroll
  for (int i = 0; i < 4; ++i) {
    const float cb = b1[(wave * 4 + i) * 16 + n];
#pragma unroll
    for (int v = 0; v < 8; ++v) acc[i][v] += cb;
  }

  // row max: 4 local tiles -> 16-lane shfl group -> cross-wave via LDS
  float rmax[8];
#pragma unroll
  for (int v = 0; v < 8; ++v) {
    float m = acc[0][v];
#pragma unroll
    for (int i = 1; i < 4; ++i) m = fmaxf(m, acc[i][v]);
    m = fmaxf(m, __shfl_xor(m, 1));
    m = fmaxf(m, __shfl_xor(m, 2));
    m = fmaxf(m, __shfl_xor(m, 4));
    m = fmaxf(m, __shfl_xor(m, 8));
    rmax[v] = m;
  }
  __syncthreads();
  if (n == 0)
#pragma unroll
    for (int v = 0; v < 8; ++v) red[wave][hh * 8 + v] = rmax[v];
  __syncthreads();
#pragma unroll
  for (int v = 0; v < 8; ++v) {
    float m = red[0][hh * 8 + v];
#pragma unroll
    for (int w = 1; w < 8; ++w) m = fmaxf(m, red[w][hh * 8 + v]);
    rmax[v] = m;
  }

  // exp + row sum
  float rsum[8];
#pragma unroll
  for (int v = 0; v < 8; ++v) {
    float s = 0.f;
#pragma unroll
    for (int i = 0; i < 4; ++i) {
      float e = __expf(acc[i][v] - rmax[v]);
      acc[i][v] = e;
      s += e;
    }
    s += __shfl_xor(s, 1);
    s += __shfl_xor(s, 2);
    s += __shfl_xor(s, 4);
    s += __shfl_xor(s, 8);
    rsum[v] = s;
  }
  __syncthreads();
  if (n == 0)
#pragma unroll
    for (int v = 0; v < 8; ++v) red[wave][hh * 8 + v] = rsum[v];
  __syncthreads();
#pragma unroll
  for (int v = 0; v < 8; ++v) {
    float s = 0.f;
#pragma unroll
    for (int w = 0; w < 8; ++w) s += red[w][hh * 8 + v];
    rsum[v] = 1.f / s;
  }

  // gatedS[b][s][e] = inputs[b][s][e] * prob[e][s]  (N-major for xw GEMM)
#pragma unroll
  for (int i = 0; i < 4; ++i) {
    const int sc = (wave * 4 + i) * 16 + n;
#pragma unroll
    for (int v = 0; v < 8; ++v) {
      const int e = e0 + hh * 8 + v;
      const float p = acc[i][v] * rsum[v];
      const float g = inputs[((size_t)b * SDIM + sc) * EDIM + e] * p;
      gatedS[((size_t)b * SDIM + sc) * EDIM + e] = (_Float16)g;
    }
  }
}

// ---------------- xw GEMM: xwT[dir,b] = kT_pad[UP,K] x Bt[b][S][K]^T ----------------
// Bt is N-major ([S][K] per batch); pipelined K-loop. Bandwidth-bound on Bt stream.

__global__ __launch_bounds__(256) void k_xw(
    const _Float16* __restrict__ AT,   // [2][UP][K]
    const _Float16* __restrict__ Bt,   // [B][S][K]  (N-major)
    float* __restrict__ CT,            // [2][B][UP][S]
    int K) {
  const int wave  = threadIdx.x >> 5;
  const int lane  = threadIdx.x & 31;
  const int ntile = blockIdx.x * 8 + wave;     // 0..31
  const int row0  = blockIdx.y * 16;           // 0/16/32
  const int dir   = blockIdx.z >> 7;
  const int b     = blockIdx.z & (BDIM - 1);

  const _Float16* A  = AT + ((size_t)dir * UP + row0) * K;
  const _Float16* Bb = Bt + ((size_t)b * SDIM + ntile * 16) * K;

  v8f acc;
#pragma unroll
  for (int v = 0; v < 8; ++v) acc[v] = 0.f;

  v16h a  = load_a_frag(A, K, lane);
  v16h bf = load_bT_frag(Bb, K, lane);

  for (int k0 = 0; k0 < K; k0 += 32) {
    const int kn = (k0 + 32 < K) ? (k0 + 32) : k0;
    v16h an = load_a_frag(A + kn, K, lane);
    v16h bn = load_bT_frag(Bb + kn, K, lane);
    acc = __builtin_amdgcn_wmma_f32_16x16x32_f16(false, a, false, bf,
                                                 (short)0, acc, false, false);
    a = an;
    bf = bn;
  }
  const int n = lane & 15, hh = lane >> 4;
  float* Cp = CT + (((size_t)dir * BDIM + b) * UP + row0) * SDIM + ntile * 16;
#pragma unroll
  for (int v = 0; v < 8; ++v) Cp[(size_t)(hh * 8 + v) * SDIM + n] = acc[v];
}

// ---------------- sequential LSTM scan: lane = batch, Wr/bias in LDS ----------------

__global__ __launch_bounds__(128) void k_scan(
    const float* __restrict__ xwT,     // [2][B][UP][S]
    const float* __restrict__ Wrf, const float* __restrict__ Wrb,   // [U][4U]
    const float* __restrict__ bvf, const float* __restrict__ bvb,   // [4U]
    _Float16* __restrict__ hseqS,      // [B][S][KP2] or nullptr
    float* __restrict__ hfinal) {      // [B][2U] or nullptr
  __shared__ float Wr[UDIM * 4 * UDIM];
  __shared__ float bz[4 * UDIM];
  const int dir = blockIdx.x;
  const float* Wrp = dir ? Wrb : Wrf;
  const float* bp  = dir ? bvb : bvf;
  for (int i = threadIdx.x; i < UDIM * 4 * UDIM; i += 128) Wr[i] = Wrp[i];
  for (int i = threadIdx.x; i < 4 * UDIM; i += 128) bz[i] = bp[i];
  __syncthreads();

  const int b = threadIdx.x;
  const float* xw = xwT + ((size_t)dir * BDIM + b) * UP * SDIM;
  float h[UDIM], c[UDIM];
#pragma unroll
  for (int u = 0; u < UDIM; ++u) { h[u] = 0.f; c[u] = 0.f; }

  for (int t = 0; t < SDIM; ++t) {
    const int tt = dir ? (SDIM - 1 - t) : t;
    float z[4 * UDIM];
#pragma unroll
    for (int j = 0; j < 4 * UDIM; ++j) z[j] = xw[(size_t)j * SDIM + tt] + bz[j];
#pragma unroll
    for (int u = 0; u < UDIM; ++u) {
      const float hu = h[u];
#pragma unroll
      for (int j = 0; j < 4 * UDIM; ++j) z[j] = fmaf(hu, Wr[u * 4 * UDIM + j], z[j]);
    }
#pragma unroll
    for (int u = 0; u < UDIM; ++u) {
      const float ig = 1.f / (1.f + __expf(-z[u]));
      const float fg = 1.f / (1.f + __expf(-z[UDIM + u]));
      const float gg = tanhf(z[2 * UDIM + u]);
      const float og = 1.f / (1.f + __expf(-z[3 * UDIM + u]));
      c[u] = fg * c[u] + ig * gg;
      h[u] = og * tanhf(c[u]);
    }
    if (hseqS) {
      // 10 contiguous halves per step (N-major layout for layer-2 B-fragments)
#pragma unroll
      for (int u = 0; u < UDIM; ++u)
        hseqS[((size_t)b * SDIM + tt) * KP2 + dir * UDIM + u] = (_Float16)h[u];
    }
  }
  if (hfinal) {
#pragma unroll
    for (int u = 0; u < UDIM; ++u) hfinal[b * 2 * UDIM + dir * UDIM + u] = h[u];
  }
}

// ---------------- head: h2 @ W3 + b3, softmax over C=32 lanes ----------------

__global__ __launch_bounds__(32) void k_head(const float* __restrict__ h2,
                                             const float* __restrict__ W3,
                                             const float* __restrict__ b3,
                                             float* __restrict__ out) {
  const int b = blockIdx.x;
  const int cI = threadIdx.x;
  float l = b3[cI];
#pragma unroll
  for (int u = 0; u < 2 * UDIM; ++u) l = fmaf(h2[b * 2 * UDIM + u], W3[u * CDIM + cI], l);
  float m = l;
  for (int off = 16; off > 0; off >>= 1) m = fmaxf(m, __shfl_xor(m, off));
  const float e = __expf(l - m);
  float s = e;
  for (int off = 16; off > 0; off >>= 1) s += __shfl_xor(s, off);
  out[b * CDIM + cI] = e / s;
}

// ---------------- launch ----------------

extern "C" void kernel_launch(void* const* d_in, const int* in_sizes, int n_in,
                              void* d_out, int out_size, void* d_ws, size_t ws_size,
                              hipStream_t stream) {
  (void)in_sizes; (void)n_in; (void)out_size; (void)ws_size;
  const float* inputs = (const float*)d_in[0];
  const float* W1  = (const float*)d_in[1];
  const float* b1  = (const float*)d_in[2];
  const float* k1f = (const float*)d_in[3];
  const float* r1f = (const float*)d_in[4];
  const float* b1f = (const float*)d_in[5];
  const float* k1b = (const float*)d_in[6];
  const float* r1b = (const float*)d_in[7];
  const float* b1b = (const float*)d_in[8];
  const float* k2f = (const float*)d_in[9];
  const float* r2f = (const float*)d_in[10];
  const float* b2f = (const float*)d_in[11];
  const float* k2b = (const float*)d_in[12];
  const float* r2b = (const float*)d_in[13];
  const float* b2b = (const float*)d_in[14];
  const float* W3  = (const float*)d_in[15];
  const float* b3  = (const float*)d_in[16];
  float* out = (float*)d_out;

  char* ws = (char*)d_ws;
  size_t off = 0;
  auto carve = [&](size_t bytes) -> void* {
    void* p = ws + off;
    off += (bytes + 255) & ~(size_t)255;
    return p;
  };
  _Float16* atP    = (_Float16*)carve((size_t)BDIM * EDIM * SDIM * sizeof(_Float16));
  _Float16* gatedS = (_Float16*)carve((size_t)BDIM * SDIM * EDIM * sizeof(_Float16));
  _Float16* w1P    = (_Float16*)carve((size_t)SDIM * SDIM * sizeof(_Float16));
  _Float16* k1T    = (_Float16*)carve((size_t)2 * UP * EDIM * sizeof(_Float16));
  _Float16* k2T    = (_Float16*)carve((size_t)2 * UP * KP2 * sizeof(_Float16));
  float*    xw1T   = (float*)carve((size_t)2 * BDIM * UP * SDIM * sizeof(float));
  _Float16* h1S    = (_Float16*)carve((size_t)BDIM * SDIM * KP2 * sizeof(_Float16));
  float*    xw2T   = (float*)carve((size_t)2 * BDIM * UP * SDIM * sizeof(float));
  float*    h2     = (float*)carve((size_t)BDIM * 2 * UDIM * sizeof(float));

  k_pack_w1<<<(NTT * NKT * 32 * 16 + 255) / 256, 256, 0, stream>>>(W1, w1P);
  k_pack_k<<<(2 * UP * EDIM + 2 * UP * KP2 + 255) / 256, 256, 0, stream>>>(
      k1f, k1b, k2f, k2b, k1T, k2T);
  {
    size_t N = (size_t)BDIM * NET * NKT * 32 * 16;
    k_pack_a<<<(unsigned)((N + 255) / 256), 256, 0, stream>>>(inputs, atP);
  }
  k_zero_h1<<<(BDIM * SDIM * KP2 + 255) / 256, 256, 0, stream>>>(h1S);

  k_attn<<<dim3(NET, BDIM), 256, 0, stream>>>(inputs, atP, w1P, b1, gatedS);
  k_xw<<<dim3(4, 3, 2 * BDIM), 256, 0, stream>>>(k1T, gatedS, xw1T, EDIM);
  k_scan<<<2, 128, 0, stream>>>(xw1T, r1f, r1b, b1f, b1b, h1S, nullptr);
  k_xw<<<dim3(4, 3, 2 * BDIM), 256, 0, stream>>>(k2T, h1S, xw2T, KP2);
  k_scan<<<2, 128, 0, stream>>>(xw2T, r2f, r2b, b2f, b2b, nullptr, h2);
  k_head<<<BDIM, 32, 0, stream>>>(h2, W3, b3, out);
}
